// CT_MHSA_38139309588872
// MI455X (gfx1250) — matrix operand order repaired and checked by reading, not compiled, not run on hardware
//
#include <hip/hip_runtime.h>

typedef __attribute__((ext_vector_type(16))) _Float16 v16h;
typedef __attribute__((ext_vector_type(4)))  _Float16 v4h;
typedef __attribute__((ext_vector_type(8)))  float    v8f;

#define D_MODEL 512
#define NTOK    2048
#define NHEAD   8
#define DH      64
#define NBATCH  4

union V16 {
    v16h v;
    uint4 q[2];
};

// A-fragment (16x32 f16) from row-major operand; base -> [row][k0].
__device__ inline v16h load_a_frag(const _Float16* base, int g) {
    V16 t;
    t.q[0] = *(const uint4*)(base + 16 * g);
    t.q[1] = *(const uint4*)(base + 16 * g + 8);
    return t.v;
}

// B-fragment (32x16 f16) from K-contiguous operand; p -> [col][k0 + 16g].
__device__ inline v16h load_b_frag(const _Float16* p) {
    V16 t;
    t.q[0] = *(const uint4*)(p);
    t.q[1] = *(const uint4*)(p + 8);
    return t.v;
}

// ---------------------------------------------------------------------------
// fp32 -> fp16 conversion, 4 elements/thread
// ---------------------------------------------------------------------------
__global__ __launch_bounds__(256) void cvt_f32_to_f16_v4(const float* __restrict__ src,
                                                         _Float16* __restrict__ dst, int n4) {
    int i = blockIdx.x * blockDim.x + threadIdx.x;
    if (i < n4) {
        float4 f = ((const float4*)src)[i];
        v4h o;
        o.x = (_Float16)f.x; o.y = (_Float16)f.y;
        o.z = (_Float16)f.z; o.w = (_Float16)f.w;
        *(v4h*)(dst + 4 * i) = o;
    }
}

// fp32 [K][N] row-major -> fp16 [N][K]
__global__ __launch_bounds__(256) void cvt_transpose_f32_to_f16(const float* __restrict__ src,
                                                                _Float16* __restrict__ dst) {
    int i = blockIdx.x * blockDim.x + threadIdx.x;
    int k = i >> 9, n = i & 511;
    dst[n * D_MODEL + k] = (_Float16)src[k * D_MODEL + n];
}

// ---------------------------------------------------------------------------
// Fused QKV projection, 32(M) x 64(N) tile per wave, k-loop step 32.
// Weights pre-transposed [n][k]. Q,K -> [b][h][n][dh]; V -> [b][h][dh][n].
// ---------------------------------------------------------------------------
__global__ __launch_bounds__(256) void qkv_gemm(const _Float16* __restrict__ xh,
                                                const _Float16* __restrict__ wqt,
                                                const _Float16* __restrict__ wkt,
                                                const _Float16* __restrict__ wvt,
                                                const float* __restrict__ bQ,
                                                const float* __restrict__ bK,
                                                const float* __restrict__ bV,
                                                _Float16* __restrict__ Qh,
                                                _Float16* __restrict__ Kh,
                                                _Float16* __restrict__ Vt) {
    const int wave = blockIdx.x * 8 + (threadIdx.x >> 5);
    const int lane = threadIdx.x & 31;
    const int g  = lane >> 4;
    const int ln = lane & 15;

    const int mat = wave / (256 * 8);     // 0=Q 1=K 2=V
    const int rem = wave % (256 * 8);
    const int mt = rem >> 3;              // 32-row tiles over 8192 rows
    const int nt = rem & 7;
    const int m0 = mt * 32;
    const int n0 = nt * 64;

    const _Float16* Wt   = (mat == 0) ? wqt : (mat == 1) ? wkt : wvt;
    const float*    bias = (mat == 0) ? bQ  : (mat == 1) ? bK  : bV;

    v8f accA[4], accB[4];
#pragma unroll
    for (int nb = 0; nb < 4; ++nb) {
        accA[nb] = (v8f){0,0,0,0,0,0,0,0};
        accB[nb] = (v8f){0,0,0,0,0,0,0,0};
    }

    const _Float16* arow0 = xh + (size_t)(m0 + ln) * D_MODEL;
    const _Float16* arow1 = xh + (size_t)(m0 + 16 + ln) * D_MODEL;

    for (int k0 = 0; k0 < D_MODEL; k0 += 32) {
        // Batch all loads of this k-step before any WMMA consumes them.
        v16h a0 = load_a_frag(arow0 + k0, g);
        v16h a1 = load_a_frag(arow1 + k0, g);
        v16h bf[4];
#pragma unroll
        for (int nb = 0; nb < 4; ++nb)
            bf[nb] = load_b_frag(Wt + (size_t)(n0 + nb * 16 + ln) * D_MODEL + k0 + 16 * g);
#pragma unroll
        for (int nb = 0; nb < 4; ++nb) {
            accA[nb] = __builtin_amdgcn_wmma_f32_16x16x32_f16(false, a0, false, bf[nb],
                                                              (short)0, accA[nb], false, false);
            accB[nb] = __builtin_amdgcn_wmma_f32_16x16x32_f16(false, a1, false, bf[nb],
                                                              (short)0, accB[nb], false, false);
        }
    }

#pragma unroll
    for (int nb = 0; nb < 4; ++nb) {
        int f = n0 + nb * 16 + ln;
        int h = f >> 6, dhi = f & 63;
        float bv = bias[f];
#pragma unroll
        for (int half = 0; half < 2; ++half) {
            const v8f& c = half ? accB[nb] : accA[nb];
#pragma unroll
            for (int r = 0; r < 8; ++r) {
                int row = m0 + half * 16 + r + 8 * g;
                int b = row >> 11, n = row & (NTOK - 1);
                _Float16 val = (_Float16)(c[r] + bv);
                if (mat == 2) {
                    Vt[(((size_t)(b * NHEAD + h) * DH) + dhi) * NTOK + n] = val;
                } else {
                    _Float16* Out = (mat == 0) ? Qh : Kh;
                    Out[(((size_t)(b * NHEAD + h) * NTOK) + n) * DH + dhi] = val;
                }
            }
        }
    }
}

// ---------------------------------------------------------------------------
// Flash attention with additive position bias P.
// Block = (b, h, 128-row super-tile); 8 independent waves, each 16 query rows.
// Per jt: all K/V/P loads issued up front; V loads and P loads hide behind
// the QK^T WMMAs and softmax VALU work.
// ---------------------------------------------------------------------------
__global__ __launch_bounds__(256) void attn_kernel(const _Float16* __restrict__ Qh,
                                                   const _Float16* __restrict__ Kh,
                                                   const _Float16* __restrict__ Vt,
                                                   const float* __restrict__ P,
                                                   _Float16* __restrict__ AO) {
    __shared__ _Float16 sP[8][16 * 32];    // per-wave prob staging (8 KB)

    const int w    = threadIdx.x >> 5;
    const int lane = threadIdx.x & 31;
    const int g  = lane >> 4;
    const int ln = lane & 15;

    const int bid = blockIdx.x;            // 0..511
    const int rt = bid & 15;
    const int h  = (bid >> 4) & 7;
    const int b  = bid >> 7;
    const int m0 = rt * 128 + w * 16;

    const _Float16* Qb = Qh + (size_t)(b * NHEAD + h) * NTOK * DH;
    const _Float16* Kb = Kh + (size_t)(b * NHEAD + h) * NTOK * DH;
    const _Float16* Vb = Vt + (size_t)(b * NHEAD + h) * DH * NTOK;
    const float*    Pb = P  + (size_t)h * NTOK * NTOK;

    const _Float16* qrow = Qb + (size_t)(m0 + ln) * DH;
    v16h qa0 = load_a_frag(qrow, g);
    v16h qa1 = load_a_frag(qrow + 32, g);

    v8f o0 = {0,0,0,0,0,0,0,0};
    v8f o1 = o0, o2 = o0, o3 = o0;
    float m_run[8], l_run[8];
#pragma unroll
    for (int r = 0; r < 8; ++r) { m_run[r] = -1e30f; l_run[r] = 0.0f; }

    const float inv_scale = 0.125f;        // 1/sqrt(64)
    _Float16* myP = &sP[w][0];

    for (int jt = 0; jt < NTOK / 32; ++jt) {
        const int j0 = jt * 32;

        // Prefetch the P bias stream two tiles ahead (true HBM-bound stream).
        if (jt + 2 < NTOK / 32) {
            __builtin_prefetch(Pb + (size_t)(m0 + 8 * g) * NTOK + j0 + 64, 0, 0);
        }

        // ---- issue ALL loads for this iteration up front ----
        v16h kb00, kb01, kb10, kb11;       // K^T fragments (S-GEMM)
        {
            const _Float16* kcol0 = Kb + (size_t)(j0 + ln) * DH;
            const _Float16* kcol1 = Kb + (size_t)(j0 + 16 + ln) * DH;
            kb00 = load_b_frag(kcol0 + 16 * g);
            kb01 = load_b_frag(kcol0 + 32 + 16 * g);
            kb10 = load_b_frag(kcol1 + 16 * g);
            kb11 = load_b_frag(kcol1 + 32 + 16 * g);
        }
        v16h vb[4];                        // V fragments (O-GEMM), needed last
#pragma unroll
        for (int nb = 0; nb < 4; ++nb)
            vb[nb] = load_b_frag(Vb + (size_t)(nb * 16 + ln) * NTOK + j0 + 16 * g);
        float pb0[8], pb1[8];              // P bias values
#pragma unroll
        for (int r = 0; r < 8; ++r) {
            int row = m0 + r + 8 * g;
            pb0[r] = Pb[(size_t)row * NTOK + (j0 + ln)];
            pb1[r] = Pb[(size_t)row * NTOK + (j0 + 16 + ln)];
        }

        // ---- S = Q K^T (16 x 32) ----
        v8f s0 = {0,0,0,0,0,0,0,0};
        s0 = __builtin_amdgcn_wmma_f32_16x16x32_f16(false, qa0, false, kb00,
                                                    (short)0, s0, false, false);
        s0 = __builtin_amdgcn_wmma_f32_16x16x32_f16(false, qa1, false, kb01,
                                                    (short)0, s0, false, false);
        v8f s1 = {0,0,0,0,0,0,0,0};
        s1 = __builtin_amdgcn_wmma_f32_16x16x32_f16(false, qa0, false, kb10,
                                                    (short)0, s1, false, false);
        s1 = __builtin_amdgcn_wmma_f32_16x16x32_f16(false, qa1, false, kb11,
                                                    (short)0, s1, false, false);

        // ---- scale + position bias, online softmax ----
        float alpha[8];
#pragma unroll
        for (int r = 0; r < 8; ++r) {
            s0[r] = s0[r] * inv_scale + pb0[r];
            s1[r] = s1[r] * inv_scale + pb1[r];
            float mx = fmaxf(s0[r], s1[r]);
#pragma unroll
            for (int off = 8; off >= 1; off >>= 1)
                mx = fmaxf(mx, __shfl_xor(mx, off, 32));
            float mnew = fmaxf(m_run[r], mx);
            alpha[r] = __expf(m_run[r] - mnew);
            m_run[r] = mnew;
            float p0 = __expf(s0[r] - mnew);
            float p1 = __expf(s1[r] - mnew);
            s0[r] = p0; s1[r] = p1;
            float rs = p0 + p1;
#pragma unroll
            for (int off = 8; off >= 1; off >>= 1)
                rs += __shfl_xor(rs, off, 32);
            l_run[r] = l_run[r] * alpha[r] + rs;
        }
#pragma unroll
        for (int r = 0; r < 8; ++r) {
            o0[r] *= alpha[r]; o1[r] *= alpha[r];
            o2[r] *= alpha[r]; o3[r] *= alpha[r];
        }

        // ---- stage probs to per-wave LDS (C-layout -> row-major), read back as A-frag ----
#pragma unroll
        for (int r = 0; r < 8; ++r) {
            int M = r + 8 * g;
            myP[M * 32 + ln]      = (_Float16)s0[r];
            myP[M * 32 + 16 + ln] = (_Float16)s1[r];
        }
        asm volatile("s_wait_dscnt 0" ::: "memory");
        __builtin_amdgcn_wave_barrier();

        V16 pt;
        pt.q[0] = *(const uint4*)(myP + ln * 32 + 16 * g);
        pt.q[1] = *(const uint4*)(myP + ln * 32 + 16 * g + 8);
        v16h pa = pt.v;

        // ---- O += probs @ V ----
        o0 = __builtin_amdgcn_wmma_f32_16x16x32_f16(false, pa, false, vb[0],
                                                    (short)0, o0, false, false);
        o1 = __builtin_amdgcn_wmma_f32_16x16x32_f16(false, pa, false, vb[1],
                                                    (short)0, o1, false, false);
        o2 = __builtin_amdgcn_wmma_f32_16x16x32_f16(false, pa, false, vb[2],
                                                    (short)0, o2, false, false);
        o3 = __builtin_amdgcn_wmma_f32_16x16x32_f16(false, pa, false, vb[3],
                                                    (short)0, o3, false, false);
    }

    // finalize: divide by row sums, store f16 to AO[b][n][h*64+dh]
#pragma unroll
    for (int r = 0; r < 8; ++r) {
        float inv = 1.0f / l_run[r];
        int row = m0 + r + 8 * g;
        _Float16* dst = AO + ((size_t)b * NTOK + row) * D_MODEL + h * DH;
        dst[0 * 16 + ln] = (_Float16)(o0[r] * inv);
        dst[1 * 16 + ln] = (_Float16)(o1[r] * inv);
        dst[2 * 16 + ln] = (_Float16)(o2[r] * inv);
        dst[3 * 16 + ln] = (_Float16)(o3[r] * inv);
    }
}

// ---------------------------------------------------------------------------
// Output projection: out = AO @ WO + bO (f32). 32x64 tile per wave.
// ---------------------------------------------------------------------------
__global__ __launch_bounds__(256) void out_gemm(const _Float16* __restrict__ AO,
                                                const _Float16* __restrict__ wot,
                                                const float* __restrict__ bO,
                                                float* __restrict__ out) {
    const int wave = blockIdx.x * 8 + (threadIdx.x >> 5);
    const int lane = threadIdx.x & 31;
    const int g  = lane >> 4;
    const int ln = lane & 15;
    const int mt = wave >> 3;             // 0..255
    const int nt = wave & 7;
    const int m0 = mt * 32;
    const int n0 = nt * 64;

    v8f accA[4], accB[4];
#pragma unroll
    for (int nb = 0; nb < 4; ++nb) {
        accA[nb] = (v8f){0,0,0,0,0,0,0,0};
        accB[nb] = (v8f){0,0,0,0,0,0,0,0};
    }

    const _Float16* arow0 = AO + (size_t)(m0 + ln) * D_MODEL;
    const _Float16* arow1 = AO + (size_t)(m0 + 16 + ln) * D_MODEL;

    for (int k0 = 0; k0 < D_MODEL; k0 += 32) {
        v16h a0 = load_a_frag(arow0 + k0, g);
        v16h a1 = load_a_frag(arow1 + k0, g);
        v16h bf[4];
#pragma unroll
        for (int nb = 0; nb < 4; ++nb)
            bf[nb] = load_b_frag(wot + (size_t)(n0 + nb * 16 + ln) * D_MODEL + k0 + 16 * g);
#pragma unroll
        for (int nb = 0; nb < 4; ++nb) {
            accA[nb] = __builtin_amdgcn_wmma_f32_16x16x32_f16(false, a0, false, bf[nb],
                                                              (short)0, accA[nb], false, false);
            accB[nb] = __builtin_amdgcn_wmma_f32_16x16x32_f16(false, a1, false, bf[nb],
                                                              (short)0, accB[nb], false, false);
        }
    }

#pragma unroll
    for (int nb = 0; nb < 4; ++nb) {
        int f = n0 + nb * 16 + ln;
        float bv = bO[f];
#pragma unroll
        for (int half = 0; half < 2; ++half) {
            const v8f& c = half ? accB[nb] : accA[nb];
#pragma unroll
            for (int r = 0; r < 8; ++r) {
                int row = m0 + half * 16 + r + 8 * g;
                out[(size_t)row * D_MODEL + f] = c[r] + bv;
            }
        }
    }
}

// ---------------------------------------------------------------------------
extern "C" void kernel_launch(void* const* d_in, const int* in_sizes, int n_in,
                              void* d_out, int out_size, void* d_ws, size_t ws_size,
                              hipStream_t stream) {
    const float* x  = (const float*)d_in[0];
    const float* WQ = (const float*)d_in[1];
    const float* bQ = (const float*)d_in[2];
    const float* WK = (const float*)d_in[3];
    const float* bK = (const float*)d_in[4];
    const float* WV = (const float*)d_in[5];
    const float* bV = (const float*)d_in[6];
    const float* P  = (const float*)d_in[7];
    const float* WO = (const float*)d_in[8];
    const float* bO = (const float*)d_in[9];
    float* out = (float*)d_out;

    char* ws = (char*)d_ws;
    const size_t XSZ = (size_t)NBATCH * NTOK * D_MODEL;     // 4,194,304 elems
    const size_t WSZ = (size_t)D_MODEL * D_MODEL;           // 262,144 elems
    size_t off = 0;
    _Float16* xh  = (_Float16*)(ws + off); off += XSZ * 2;  // 8 MB
    _Float16* wqt = (_Float16*)(ws + off); off += WSZ * 2;
    _Float16* wkt = (_Float16*)(ws + off); off += WSZ * 2;
    _Float16* wvt = (_Float16*)(ws + off); off += WSZ * 2;
    _Float16* wot = (_Float16*)(ws + off); off += WSZ * 2;
    _Float16* Qh  = (_Float16*)(ws + off); off += XSZ * 2;
    _Float16* Kh  = (_Float16*)(ws + off); off += XSZ * 2;
    _Float16* Vt  = (_Float16*)(ws + off); off += XSZ * 2;
    _Float16* AOh = (_Float16*)(ws + off); off += XSZ * 2;  // total ~42 MB

    cvt_f32_to_f16_v4<<<(int)((XSZ / 4 + 255) / 256), 256, 0, stream>>>(x, xh, (int)(XSZ / 4));
    cvt_transpose_f32_to_f16<<<(int)((WSZ + 255) / 256), 256, 0, stream>>>(WQ, wqt);
    cvt_transpose_f32_to_f16<<<(int)((WSZ + 255) / 256), 256, 0, stream>>>(WK, wkt);
    cvt_transpose_f32_to_f16<<<(int)((WSZ + 255) / 256), 256, 0, stream>>>(WV, wvt);
    cvt_transpose_f32_to_f16<<<(int)((WSZ + 255) / 256), 256, 0, stream>>>(WO, wot);

    // QKV projection: 3 * 256 * 8 = 6144 wave tiles / 8 per block
    qkv_gemm<<<768, 256, 0, stream>>>(xh, wqt, wkt, wvt, bQ, bK, bV, Qh, Kh, Vt);

    // flash attention: B*H*(N/128) = 512 blocks
    attn_kernel<<<512, 256, 0, stream>>>(Qh, Kh, Vt, P, AOh);

    // output projection: 256 * 8 = 2048 wave tiles / 8 per block
    out_gemm<<<256, 256, 0, stream>>>(AOh, wot, bO, out);
}
